// StateProjector_34754875359790
// MI455X (gfx1250) — matrix operand order repaired and checked by reading, not compiled, not run on hardware
//
#include <hip/hip_runtime.h>
#include <math.h>

#define NE 8
#define SS 64
#define HH 2048
#define DD 1024
#define BB 8192
#define BROW 72   // padded LDS row length (elements) -> 144B, conflict-free
#define NBLK 256  // block N tile
#define MBLK 64   // block M tile

typedef __attribute__((ext_vector_type(16))) __bf16 v16bf;
typedef __attribute__((ext_vector_type(8)))  __bf16 v8bf;
typedef __attribute__((ext_vector_type(8)))  float  v8f;

// ---------------------------------------------------------------- helpers
__device__ __forceinline__ float gelu_exact(float x) {
  return 0.5f * x * (1.0f + erff(x * 0.7071067811865475f));
}

__device__ __forceinline__ v16bf combine16(v8bf a0, v8bf a1) {
  v16bf r;
#pragma unroll
  for (int i = 0; i < 8; ++i) { r[i] = a0[i]; r[i + 8] = a1[i]; }
  return r;
}

// 16-bit A 16x32 fragment (ISA 7.12.2): lane L holds row M=L&15;
// lanes 0-15: K chunks [0..7],[16..23]; lanes 16-31: [8..15],[24..31]
__device__ __forceinline__ v16bf load_a_frag(const __bf16* arow, int k, int lane) {
  const int k0 = (lane >> 4) << 3;                 // 0 or 8
  v8bf a0 = *(const v8bf*)(arow + k + k0);
  v8bf a1 = *(const v8bf*)(arow + k + k0 + 16);
  return combine16(a0, a1);
}

// ---------------------------------------------------------------- routing
__global__ void zero_small_kernel(int* cnt) {
  if (threadIdx.x < NE) cnt[threadIdx.x] = 0;
}
__global__ void count_kernel(const int* __restrict__ eidx, int* __restrict__ cnt) {
  int b = blockIdx.x * 256 + threadIdx.x;
  if (b < BB) atomicAdd(&cnt[eidx[b]], 1);
}
__global__ void scan_kernel(const int* __restrict__ cnt, int* __restrict__ offs,
                            int* __restrict__ fill) {
  if (threadIdx.x == 0) {
    int a = 0;
    for (int e = 0; e < NE; ++e) { offs[e] = a; a += cnt[e]; fill[e] = 0; }
    offs[NE] = a;
  }
}
__global__ void perm_kernel(const int* __restrict__ eidx, const int* __restrict__ offs,
                            int* __restrict__ fill, int* __restrict__ perm) {
  int b = blockIdx.x * 256 + threadIdx.x;
  if (b < BB) {
    int e = eidx[b];
    int p = atomicAdd(&fill[e], 1);
    perm[offs[e] + p] = b;
  }
}
__global__ void gather_kernel(const int* __restrict__ perm, const int* __restrict__ offs,
                              const float* __restrict__ x, const int* __restrict__ hp,
                              __bf16* __restrict__ Xg, float* __restrict__ selg,
                              int* __restrict__ eofg) {
  int g = blockIdx.x, t = threadIdx.x;               // 64 threads
  int b = perm[g];
  Xg[(size_t)g * SS + t] = (__bf16)x[(size_t)b * SS + t];
  if (t == 0) {
    selg[g] = (float)hp[b];
    int e = 0;
    while (e < NE - 1 && g >= offs[e + 1]) ++e;
    eofg[g] = e;
  }
}

// -------------------------------------------------- weight transpose + cvt
// in: [E][K][N] f32 -> out: [E][N][K] bf16
__global__ __launch_bounds__(256) void transpose_cvt_kernel(
    const float* __restrict__ in, __bf16* __restrict__ out, int K, int N) {
  __shared__ float tile[32][33];
  const int e = blockIdx.z;
  const float* src = in + (size_t)e * K * N;
  __bf16* dst = out + (size_t)e * N * K;
  const int n0 = blockIdx.x * 32, k0 = blockIdx.y * 32;
  const int tx = threadIdx.x, ty = threadIdx.y;      // (32, 8)
#pragma unroll
  for (int i = 0; i < 32; i += 8)
    tile[ty + i][tx] = src[(size_t)(k0 + ty + i) * N + (n0 + tx)];
  __syncthreads();
#pragma unroll
  for (int i = 0; i < 32; i += 8)
    dst[(size_t)(n0 + ty + i) * K + (k0 + tx)] = (__bf16)tile[tx][ty + i];
}

// ---------------------------------------------------------------- GEMM
enum { EPI_GELU_BF16 = 0, EPI_MIX_STORE = 1, EPI_MIX_ADD = 2, EPI_SCATTER = 3 };

// Issue one NBLK(N) x 64(K) bf16 chunk of Bt into LDS via async DMA.
// NBLK*8 x 16B transfers; 256 threads -> 8 async instructions per thread.
__device__ __forceinline__ void async_b_chunk(const __bf16* bte, int K, int ntile_blk,
                                              int kc, unsigned ldsbase, int tid) {
#pragma unroll
  for (int it = 0; it < (NBLK * 8) / 256; ++it) {
    int idx = it * 256 + tid;                        // 0..NBLK*8-1
    int n = idx >> 3, c8 = idx & 7;
    const __bf16* gp = bte + (size_t)(ntile_blk + n) * K + kc + c8 * 8;
    unsigned l = ldsbase + (unsigned)(n * (BROW * 2) + c8 * 16);
    asm volatile("global_load_async_to_lds_b128 %0, %1, off"
                 :: "v"(l), "v"((unsigned long long)(uintptr_t)gp)
                 : "memory");
  }
}

// Block: 256 thr = 8 waves in a 2(M) x 4(N) grid of 32x64 wave tiles.
// Block tile: 64 rows x 256 cols. B double-buffered in LDS via async DMA.
// Wave tile 32x64: 2 A-frags x 4 B-frags -> 8 WMMAs; B reused 2x from regs.
template <int EPI>
__global__ __launch_bounds__(256) void gemm_bf16_kernel(
    const __bf16* __restrict__ A, int lda,
    const __bf16* __restrict__ Bt, int K,
    const float* __restrict__ bias, int N,
    void* __restrict__ outp, int ldout,
    const int* __restrict__ counts, const int* __restrict__ offs,
    const float* __restrict__ selg, const int* __restrict__ perm) {
  __shared__ __bf16 Bs[2][NBLK][BROW];

  const int e  = blockIdx.z;
  const int ne = counts[e];
  if ((int)blockIdx.y * MBLK >= ne) return;          // uniform block exit
  const int tid  = threadIdx.x;
  const int wave = tid >> 5;
  const int lane = tid & 31;
  const int wm = wave >> 2, wn = wave & 3;
  const int mtile     = blockIdx.y * MBLK + wm * 32;
  const int ntile_blk = blockIdx.x * NBLK;
  const int base = offs[e];

  int mrow0 = mtile + (lane & 15);
  int mrow1 = mrow0 + 16;
  if (mrow0 >= ne) mrow0 = ne - 1;                   // clamp loads, EXEC stays full
  if (mrow1 >= ne) mrow1 = ne - 1;
  const __bf16* arow0 = A + (size_t)(base + mrow0) * lda;
  const __bf16* arow1 = A + (size_t)(base + mrow1) * lda;
  const __bf16* bte  = Bt + (size_t)e * N * K;
  const float*  be   = bias + (size_t)e * N;

  const unsigned lds0 = (unsigned)(uintptr_t)(&Bs[0][0][0]);
  const unsigned lbase[2] = {lds0, lds0 + (unsigned)NBLK * (BROW * 2)};
  const int k0b = (lane >> 4) << 3;                  // B K-chunk select (0 or 8)

  v8f acc[2][4] = {};

  auto compute_chunk = [&](int kc, int p) {
#pragma unroll
    for (int kk = 0; kk < 64; kk += 32) {
      v16bf af0 = load_a_frag(arow0, kc + kk, lane);
      v16bf af1 = load_a_frag(arow1, kc + kk, lane);
      v16bf bf[4];
#pragma unroll
      for (int ni = 0; ni < 4; ++ni) {
        const __bf16* bp = &Bs[p][wn * 64 + ni * 16 + (lane & 15)][kk + k0b];
        bf[ni] = combine16(*(const v8bf*)bp, *(const v8bf*)(bp + 16));
      }
#pragma unroll
      for (int ni = 0; ni < 4; ++ni)
        acc[0][ni] = __builtin_amdgcn_wmma_f32_16x16x32_bf16(
            false, af0, false, bf[ni], (short)0, acc[0][ni], false, false);
#pragma unroll
      for (int ni = 0; ni < 4; ++ni)
        acc[1][ni] = __builtin_amdgcn_wmma_f32_16x16x32_bf16(
            false, af1, false, bf[ni], (short)0, acc[1][ni], false, false);
    }
  };

  async_b_chunk(bte, K, ntile_blk, 0, lbase[0], tid);
  int p = 0;
  int kc = 0;
  for (; kc < K - 64; kc += 64) {                    // steady state, branch-free
    async_b_chunk(bte, K, ntile_blk, kc + 64, lbase[p ^ 1], tid);
    asm volatile("s_wait_asynccnt 0x8" ::: "memory");
    __syncthreads();
    compute_chunk(kc, p);
    __syncthreads();
    p ^= 1;
  }
  asm volatile("s_wait_asynccnt 0x0" ::: "memory");  // tail chunk
  __syncthreads();
  compute_chunk(kc, p);

  // C/D layout: VGPR r -> M = r (lanes 0-15) or r+8 (lanes 16-31); N = lane&15
  const int mhalf = (lane >> 4) * 8;
#pragma unroll
  for (int mi = 0; mi < 2; ++mi) {
#pragma unroll
    for (int ni = 0; ni < 4; ++ni) {
      const int col = ntile_blk + wn * 64 + ni * 16 + (lane & 15);
#pragma unroll
      for (int r = 0; r < 8; ++r) {
        const int mo = mtile + mi * 16 + r + mhalf;
        if (mo >= ne) continue;
        const int g = base + mo;
        const float v = acc[mi][ni][r] + be[col];
        if (EPI == EPI_GELU_BF16) {
          ((__bf16*)outp)[(size_t)g * ldout + col] = (__bf16)gelu_exact(v);
        } else if (EPI == EPI_MIX_STORE) {
          ((float*)outp)[(size_t)g * ldout + col] = v * (1.0f - selg[g]);
        } else if (EPI == EPI_MIX_ADD) {
          float* o = (float*)outp + (size_t)g * ldout + col;
          *o = *o + v * selg[g];
        } else {  // EPI_SCATTER: back to original row order
          ((float*)outp)[(size_t)perm[g] * ldout + col] = v;
        }
      }
    }
  }
}

// ---------------------------------------------------------------- LayerNorm
__global__ __launch_bounds__(256) void ln_kernel(
    const float* __restrict__ mixed, __bf16* __restrict__ nrm,
    const float* __restrict__ lng, const float* __restrict__ lnb,
    const int* __restrict__ eofg) {
  const int g = blockIdx.x, t = threadIdx.x;
  const float* row = mixed + (size_t)g * DD;
  float s = 0.f, s2 = 0.f;
  for (int i = t; i < DD; i += 256) { float v = row[i]; s += v; s2 += v * v; }
  __shared__ float sh[256], sh2[256];
  sh[t] = s; sh2[t] = s2;
  __syncthreads();
  for (int w = 128; w > 0; w >>= 1) {
    if (t < w) { sh[t] += sh[t + w]; sh2[t] += sh2[t + w]; }
    __syncthreads();
  }
  const float mu  = sh[0] * (1.0f / DD);
  const float var = sh2[0] * (1.0f / DD) - mu * mu;
  const float rs  = rsqrtf(var + 1e-5f);
  const int e = eofg[g];
  const float* gg = lng + (size_t)e * DD;
  const float* bb = lnb + (size_t)e * DD;
  for (int i = t; i < DD; i += 256)
    nrm[(size_t)g * DD + i] = (__bf16)((row[i] - mu) * rs * gg[i] + bb[i]);
}

// ---------------------------------------------------------------- launch
extern "C" void kernel_launch(void* const* d_in, const int* in_sizes, int n_in,
                              void* d_out, int out_size, void* d_ws, size_t ws_size,
                              hipStream_t stream) {
  const float* raw  = (const float*)d_in[0];
  const int*   hp   = (const int*)d_in[1];
  const int*   eidx = (const int*)d_in[2];
  const float* pw1  = (const float*)d_in[3];
  const float* pb1  = (const float*)d_in[4];
  const float* pw2  = (const float*)d_in[5];
  const float* pb2  = (const float*)d_in[6];
  const float* qw1  = (const float*)d_in[7];
  const float* qb1  = (const float*)d_in[8];
  const float* qw2  = (const float*)d_in[9];
  const float* qb2  = (const float*)d_in[10];
  const float* lng  = (const float*)d_in[11];
  const float* lnb  = (const float*)d_in[12];
  const float* tw1  = (const float*)d_in[13];
  const float* tb1  = (const float*)d_in[14];
  const float* tw2  = (const float*)d_in[15];
  const float* tb2  = (const float*)d_in[16];

  char* ws = (char*)d_ws;
  size_t o = 0;
  auto alloc = [&](size_t bytes) -> void* {
    void* p = ws + o;
    o += (bytes + 255) & ~(size_t)255;
    return p;
  };
  int*    cnt   = (int*)alloc(NE * sizeof(int));
  int*    offs  = (int*)alloc((NE + 1) * sizeof(int));
  int*    fill  = (int*)alloc(NE * sizeof(int));
  int*    perm  = (int*)alloc(BB * sizeof(int));
  int*    eofg  = (int*)alloc(BB * sizeof(int));
  float*  selg  = (float*)alloc(BB * sizeof(float));
  __bf16* Xg    = (__bf16*)alloc((size_t)BB * SS * 2);
  __bf16* pw1t  = (__bf16*)alloc((size_t)NE * HH * SS * 2);
  __bf16* qw1t  = (__bf16*)alloc((size_t)NE * HH * SS * 2);
  __bf16* pw2t  = (__bf16*)alloc((size_t)NE * DD * HH * 2);
  __bf16* qw2t  = (__bf16*)alloc((size_t)NE * DD * HH * 2);
  __bf16* tw1t  = (__bf16*)alloc((size_t)NE * HH * DD * 2);
  __bf16* tw2t  = (__bf16*)alloc((size_t)NE * DD * HH * 2);
  __bf16* Hp    = (__bf16*)alloc((size_t)BB * HH * 2);
  __bf16* Hq    = (__bf16*)alloc((size_t)BB * HH * 2);
  float*  Mixed = (float*)alloc((size_t)BB * DD * 4);
  __bf16* Nrm   = (__bf16*)alloc((size_t)BB * DD * 2);
  (void)ws_size; (void)in_sizes; (void)n_in; (void)out_size;

  // routing
  zero_small_kernel<<<1, 32, 0, stream>>>(cnt);
  count_kernel<<<BB / 256, 256, 0, stream>>>(eidx, cnt);
  scan_kernel<<<1, 32, 0, stream>>>(cnt, offs, fill);
  perm_kernel<<<BB / 256, 256, 0, stream>>>(eidx, offs, fill, perm);
  gather_kernel<<<BB, SS, 0, stream>>>(perm, offs, raw, hp, Xg, selg, eofg);

  // weight transpose + fp32 -> bf16
  dim3 tb(32, 8);
  transpose_cvt_kernel<<<dim3(HH / 32, SS / 32, NE), tb, 0, stream>>>(pw1, pw1t, SS, HH);
  transpose_cvt_kernel<<<dim3(HH / 32, SS / 32, NE), tb, 0, stream>>>(qw1, qw1t, SS, HH);
  transpose_cvt_kernel<<<dim3(DD / 32, HH / 32, NE), tb, 0, stream>>>(pw2, pw2t, HH, DD);
  transpose_cvt_kernel<<<dim3(DD / 32, HH / 32, NE), tb, 0, stream>>>(qw2, qw2t, HH, DD);
  transpose_cvt_kernel<<<dim3(HH / 32, DD / 32, NE), tb, 0, stream>>>(tw1, tw1t, DD, HH);
  transpose_cvt_kernel<<<dim3(DD / 32, HH / 32, NE), tb, 0, stream>>>(tw2, tw2t, HH, DD);

  // per-expert WMMA GEMM pipeline (block tile 64M x 256N)
  dim3 blk(256);
  dim3 g1(HH / NBLK, BB / MBLK, NE);
  dim3 g2(DD / NBLK, BB / MBLK, NE);
  gemm_bf16_kernel<EPI_GELU_BF16><<<g1, blk, 0, stream>>>(
      Xg, SS, pw1t, SS, pb1, HH, Hp, HH, cnt, offs, selg, perm);
  gemm_bf16_kernel<EPI_GELU_BF16><<<g1, blk, 0, stream>>>(
      Xg, SS, qw1t, SS, qb1, HH, Hq, HH, cnt, offs, selg, perm);
  gemm_bf16_kernel<EPI_MIX_STORE><<<g2, blk, 0, stream>>>(
      Hp, HH, pw2t, HH, pb2, DD, Mixed, DD, cnt, offs, selg, perm);
  gemm_bf16_kernel<EPI_MIX_ADD><<<g2, blk, 0, stream>>>(
      Hq, HH, qw2t, HH, qb2, DD, Mixed, DD, cnt, offs, selg, perm);
  ln_kernel<<<BB, 256, 0, stream>>>(Mixed, Nrm, lng, lnb, eofg);
  gemm_bf16_kernel<EPI_GELU_BF16><<<g1, blk, 0, stream>>>(
      Nrm, DD, tw1t, DD, tb1, HH, Hp, HH, cnt, offs, selg, perm);
  gemm_bf16_kernel<EPI_SCATTER><<<g2, blk, 0, stream>>>(
      Hp, HH, tw2t, HH, tb2, DD, d_out, DD, cnt, offs, selg, perm);
}
